// LorentzTransformerEncoder_86380382257349
// MI455X (gfx1250) — matrix operand order repaired and checked by reading, not compiled, not run on hardware
//
#include <hip/hip_runtime.h>
#include <hip/hip_bf16.h>
#include <math.h>

typedef __attribute__((ext_vector_type(16))) _Float16 v16h;
typedef __attribute__((ext_vector_type(8)))  _Float16 v8h;
typedef __attribute__((ext_vector_type(8)))  float    v8f;

#define B_   8
#define N_   577
#define D_   769
#define S_   768
#define H_   12
#define DH_  64
#define M3_  3073
#define MS_  3072
#define BN_  (B_*N_)      /* 4616 */
#define MP_  4672         /* BN padded to mult of 64 */
#define DP_  800          /* D padded to mult of 32 */
#define NPQ_ 640          /* per-head rows padded to mult of 64 */
#define NPK_ 608          /* per-head cols padded to mult of 32 */
#define FFP_ 3104         /* M3 padded to mult of 32 */
#define BH_  (B_*H_)      /* 96 */

// ---------------- CDNA5 async global->LDS (ASYNCcnt) ----------------
__device__ __forceinline__ void async_b128(unsigned lds_addr, const void* sbase, int voff) {
  asm volatile("global_load_async_to_lds_b128 %0, %1, %2"
               :: "v"(lds_addr), "v"(voff), "s"(sbase) : "memory");
}
// partial wait: allow the 3 just-issued (next-buffer) loads to stay in flight
__device__ __forceinline__ void wait_async_le3() {
  asm volatile("s_wait_asynccnt 3" ::: "memory");
}
__device__ __forceinline__ void wait_async_le0() {
  asm volatile("s_wait_asynccnt 0" ::: "memory");
}

// ---------------- reductions (wave32) ----------------
__device__ __forceinline__ float blockReduceSum(float v, float* sh) {
  for (int o = 16; o > 0; o >>= 1) v += __shfl_down(v, o, 32);
  int lane = threadIdx.x & 31, w = threadIdx.x >> 5;
  __syncthreads();
  if (lane == 0) sh[w] = v;
  __syncthreads();
  int nw = (blockDim.x + 31) >> 5;
  float r = (threadIdx.x < nw) ? sh[threadIdx.x] : 0.f;
  if (w == 0) for (int o = 16; o > 0; o >>= 1) r += __shfl_down(r, o, 32);
  if (threadIdx.x == 0) sh[0] = r;
  __syncthreads();
  return sh[0];
}
__device__ __forceinline__ float blockReduceMax(float v, float* sh) {
  for (int o = 16; o > 0; o >>= 1) v = fmaxf(v, __shfl_down(v, o, 32));
  int lane = threadIdx.x & 31, w = threadIdx.x >> 5;
  __syncthreads();
  if (lane == 0) sh[w] = v;
  __syncthreads();
  int nw = (blockDim.x + 31) >> 5;
  float r = (threadIdx.x < nw) ? sh[threadIdx.x] : -3.4e38f;
  if (w == 0) for (int o = 16; o > 0; o >>= 1) r = fmaxf(r, __shfl_down(r, o, 32));
  if (threadIdx.x == 0) sh[0] = r;
  __syncthreads();
  return sh[0];
}
__device__ __forceinline__ float softplusf(float x) {
  return (x > 20.f) ? x : log1pf(expf(x));
}

// ---------------- generic f16 WMMA GEMM: C(f32) = A(MxK) * B^T (B stored [n][k]) ---
// block = 128 threads (4 waves); block tile = 64(M) x 32(N); K tile = 32.
// Double-buffered LDS fed by async global->LDS; every thread issues exactly
// 3 async b128 loads per K step (2 A chunks + 1 B chunk) -> no divergence.
// Each wave computes a 16x32 strip of C (two 16x16 WMMA accumulators).
// All dims pre-padded: M%64==0, N%32==0, K%32==0. A zero-padded, so no guards.
__global__ __launch_bounds__(128)
void k_gemm(const _Float16* __restrict__ A, const _Float16* __restrict__ Bm,
            float* __restrict__ C, int lda, int ldb, int ldc, int K,
            long strA, long strB, long strC)
{
  __shared__ _Float16 tA[2][64 * 32];
  __shared__ _Float16 tB[2][32 * 32];
  const int tid  = threadIdx.x;
  const int lane = tid & 31;
  const int wid  = tid >> 5;            // 0..3 = M subtile
  const int g    = lane >> 4, l16 = lane & 15;
  const long z   = blockIdx.z;

  const _Float16* Abase = A + z * strA + (long)(blockIdx.x * 64) * lda;
  const _Float16* Bbase = Bm + z * strB + (long)(blockIdx.y * 32) * ldb;

  // A tile: 64 rows x 32 halfs = 256 x 16B chunks; thread does chunks tid, tid+128
  const int avoff0 = ((tid >> 2) * lda + (tid & 3) * 8) * 2;
  const int avoff1 = (((tid >> 2) + 32) * lda + (tid & 3) * 8) * 2;
  // B tile: 32 rows x 32 halfs = 128 x 16B chunks; thread does chunk tid
  const int bvoff  = ((tid >> 2) * ldb + (tid & 3) * 8) * 2;

  const unsigned aL[2] = { (unsigned)(size_t)(&tA[0][0]) + (unsigned)tid * 16u,
                           (unsigned)(size_t)(&tA[1][0]) + (unsigned)tid * 16u };
  const unsigned bL[2] = { (unsigned)(size_t)(&tB[0][0]) + (unsigned)tid * 16u,
                           (unsigned)(size_t)(&tB[1][0]) + (unsigned)tid * 16u };

  v8f acc0 = {}, acc1 = {};
  const int nk = K >> 5;

  // preload buffer 0
  {
    const _Float16* Ab = Abase;
    async_b128(aL[0],         (const void*)Ab, avoff0);
    async_b128(aL[0] + 2048u, (const void*)Ab, avoff1);
    async_b128(bL[0],         (const void*)Bbase, bvoff);
  }
  for (int i = 0; i < nk; i++) {
    const int cur = i & 1;
    if (i + 1 < nk) {
      const _Float16* Ab = Abase + (i + 1) * 32;
      const _Float16* Bb = Bbase + (i + 1) * 32;
      async_b128(aL[cur ^ 1],         (const void*)Ab, avoff0);
      async_b128(aL[cur ^ 1] + 2048u, (const void*)Ab, avoff1);
      async_b128(bL[cur ^ 1],         (const void*)Bb, bvoff);
      wait_async_le3();   // drain current buffer's 3 loads (in-order completion)
    } else {
      wait_async_le0();
    }
    __syncthreads();
    // A frag (16x32 f16): lane group g: K {16g..16g+15} of row m
    v8h a0 = *(const v8h*)&tA[cur][(wid * 16 + l16) * 32 + 16 * g];
    v8h a1 = *(const v8h*)&tA[cur][(wid * 16 + l16) * 32 + 16 * g + 8];
    // B frags (32x16 f16) for the two N subtiles: lane group g: K {16g..16g+15}
    v8h b0 = *(const v8h*)&tB[cur][l16 * 32 + 16 * g];
    v8h b1 = *(const v8h*)&tB[cur][l16 * 32 + 16 * g + 8];
    v8h c0 = *(const v8h*)&tB[cur][(16 + l16) * 32 + 16 * g];
    v8h c1 = *(const v8h*)&tB[cur][(16 + l16) * 32 + 16 * g + 8];
    v16h av, bv, cv;
#pragma unroll
    for (int t = 0; t < 8; t++) {
      av[t] = a0[t]; av[t + 8] = a1[t];
      bv[t] = b0[t]; bv[t + 8] = b1[t];
      cv[t] = c0[t]; cv[t + 8] = c1[t];
    }
    acc0 = __builtin_amdgcn_wmma_f32_16x16x32_f16(false, av, false, bv, (short)0, acc0, false, false);
    acc1 = __builtin_amdgcn_wmma_f32_16x16x32_f16(false, av, false, cv, (short)0, acc1, false, false);
    __syncthreads();
  }
  float* Cz = C + z * strC;
  const int crow0 = blockIdx.x * 64 + wid * 16 + 8 * g;
  const int ccol0 = blockIdx.y * 32 + l16;
#pragma unroll
  for (int r = 0; r < 8; r++) {
    Cz[(long)(crow0 + r) * ldc + ccol0]      = acc0[r];
    Cz[(long)(crow0 + r) * ldc + ccol0 + 16] = acc1[r];
  }
}

// ---------------- weight pad+convert f32 -> f16 ----------------
__global__ void k_padw(const float* __restrict__ W, _Float16* __restrict__ O,
                       int rows, int cols, int prows, int pcols)
{
  long i = (long)blockIdx.x * blockDim.x + threadIdx.x;
  long tot = (long)prows * pcols;
  if (i >= tot) return;
  int r = (int)(i / pcols), c = (int)(i % pcols);
  float v = (r < rows && c < cols) ? W[(long)r * cols + c] : 0.f;
  O[i] = (_Float16)v;
}

// ---------------- lorentz layernorm 1: x -> xn(f16, padded) ----------------
__global__ __launch_bounds__(256)
void k_ln1(const float* __restrict__ x, const float* __restrict__ gamma,
           const float* __restrict__ beta, _Float16* __restrict__ xnh)
{
  __shared__ float sh[16];
  int row = blockIdx.x;
  _Float16* out = xnh + (long)row * DP_;
  if (row >= BN_) { for (int c = threadIdx.x; c < DP_; c += 256) out[c] = (_Float16)0.f; return; }
  const float* xs = x + (long)row * D_ + 1;
  float s = 0.f, s2 = 0.f;
  for (int c = threadIdx.x; c < S_; c += 256) { float v = xs[c]; s += v; s2 += v * v; }
  s = blockReduceSum(s, sh);
  s2 = blockReduceSum(s2, sh);
  float mu = s / S_, var = s2 / S_ - mu * mu;
  float inv = rsqrtf(var + 1e-5f);
  float nsq = 0.f;
  for (int c = threadIdx.x; c < S_; c += 256) {
    float v = (xs[c] - mu) * inv * gamma[c] + beta[c];
    nsq += v * v;
  }
  nsq = blockReduceSum(nsq, sh);
  float t = sqrtf(nsq + 1.f);
  for (int c = threadIdx.x; c < S_; c += 256) {
    float v = (xs[c] - mu) * inv * gamma[c] + beta[c];
    out[1 + c] = (_Float16)v;
  }
  if (threadIdx.x == 0) out[0] = (_Float16)t;
  for (int c = S_ + 1 + threadIdx.x; c < DP_; c += 256) out[c] = (_Float16)0.f;
}

// ---------------- split heads, recompute times, logmap0(v) transposed ----------------
__global__ void k_heads(const float* __restrict__ q, const float* __restrict__ k,
                        const float* __restrict__ v,
                        _Float16* __restrict__ qh, _Float16* __restrict__ kh,
                        _Float16* __restrict__ vtT,
                        float* __restrict__ qt, float* __restrict__ kt)
{
  int id = blockIdx.x * blockDim.x + threadIdx.x;
  if (id >= BH_ * NPQ_) return;
  int bh = id / NPQ_, j = id % NPQ_;
  int b = bh / H_, h = bh % H_;
  bool valid = (j < N_);
  long src = valid ? ((long)(b * N_ + j) * DP_ + 1 + h * DH_) : 0;
  float sq = 0.f, sk = 0.f, sv = 0.f;
  if (valid) {
    for (int d = 0; d < DH_; d++) {
      float a = q[src + d], bb = k[src + d], c = v[src + d];
      sq += a * a; sk += bb * bb; sv += c * c;
    }
  }
  float qtv = sqrtf(sq + 1.f), ktv = sqrtf(sk + 1.f), vt = sqrtf(sv + 1.f);
  float av = fmaxf(vt, 1.f + 1e-7f);
  float coef = acoshf(av) / sqrtf(av * av - 1.f);
  _Float16* qrow = qh + ((long)bh * NPQ_ + j) * DH_;
  for (int d = 0; d < DH_; d++) qrow[d] = valid ? (_Float16)q[src + d] : (_Float16)0.f;
  qt[(long)bh * NPQ_ + j] = valid ? qtv : 1.f;
  if (j < NPK_) {
    _Float16* krow = kh + ((long)bh * NPK_ + j) * DH_;
    for (int d = 0; d < DH_; d++) krow[d] = valid ? (_Float16)k[src + d] : (_Float16)0.f;
    kt[(long)bh * NPK_ + j] = valid ? ktv : 1.f;
    _Float16* vcol = vtT + (long)bh * DH_ * NPK_ + j;
    for (int d = 0; d < DH_; d++) vcol[(long)d * NPK_] = valid ? (_Float16)(coef * v[src + d]) : (_Float16)0.f;
  }
}

// ---------------- hyperbolic score logits + softmax -> f16 scores ----------------
__global__ __launch_bounds__(256)
void k_logits(const float* __restrict__ inner, const float* __restrict__ qt,
              const float* __restrict__ kt, _Float16* __restrict__ sc,
              const float* __restrict__ temp, const float* __restrict__ araw,
              const float* __restrict__ traw, const float* __restrict__ lraw)
{
  __shared__ float sh[16];
  int i = blockIdx.x, bh = blockIdx.y;
  _Float16* srow = sc + ((long)bh * NPQ_ + i) * NPK_;
  if (i >= N_) { for (int j = threadIdx.x; j < NPK_; j += 256) srow[j] = (_Float16)0.f; return; }
  float alpha = softplusf(araw[0]);
  float tau   = softplusf(traw[0]);
  float lam   = softplusf(lraw[0]);
  float T     = temp[0];
  float qti = qt[(long)bh * NPQ_ + i];
  float ct = acoshf(fmaxf(qti, 1.f + 1e-7f));
  float Bc = alpha * ct / (1.f + alpha * ct);
  float entc = log1pf(expf(-0.1f));
  const float* irow = inner + ((long)bh * NPQ_ + i) * NPK_;
  float l[3], e[3];
  float mx = -1e30f;
#pragma unroll
  for (int t = 0; t < 3; t++) {
    int j = threadIdx.x + t * 256;
    float lv = -1e30f;
    if (j < N_) {
      float d = irow[j];
      float ktj = kt[(long)bh * NPK_ + j];
      float iq = -qti * ktj + d;                       // inner_QK
      float dist = acoshf(fmaxf(-iq, 1.f));
      float nqk = fmaxf(iq * iq - 1.f, 0.f);
      float noq = fmaxf(qti * qti - 1.f, 0.f);
      float numer = -ktj - iq * qti;
      float denom = sqrtf(fmaxf(nqk * noq, 1e-14f));
      float Z = fminf(fmaxf(numer / denom, -1.f), 1.f);
      float sd = fminf(dist, 40.f) * (1.f / 15.f);
      float lc = sd + softplusf(-2.f * sd) - 0.69314718056f;
      float ent = softplusf(Bc + Z - 0.1f) - entc;
      lv = (-lam * lc - tau * ent) / T;
    }
    l[t] = lv;
    mx = fmaxf(mx, lv);
  }
  mx = blockReduceMax(mx, sh);
  float se = 0.f;
#pragma unroll
  for (int t = 0; t < 3; t++) {
    int j = threadIdx.x + t * 256;
    float ev = (j < N_) ? expf(l[t] - mx) : 0.f;
    e[t] = ev; se += ev;
  }
  se = blockReduceSum(se, sh);
  float inv = 1.f / se;
#pragma unroll
  for (int t = 0; t < 3; t++) {
    int j = threadIdx.x + t * 256;
    if (j < NPK_) srow[j] = (j < N_) ? (_Float16)(e[t] * inv) : (_Float16)0.f;
  }
}

// ---------------- expmap0 per head + write spatial part of attn row ----------------
__global__ __launch_bounds__(64)
void k_expmap(const float* __restrict__ means, _Float16* __restrict__ attnh,
              float* __restrict__ th2)
{
  __shared__ float sh[2];
  int n = blockIdx.x, bh = blockIdx.y;
  int b = bh / H_, h = bh % H_;
  int d = threadIdx.x;
  float ms = means[((long)bh * NPQ_ + n) * DH_ + d] * (1.f / (1.f + 1e-8f));
  float sq = ms * ms;
  for (int o = 16; o > 0; o >>= 1) sq += __shfl_down(sq, o, 32);
  if ((threadIdx.x & 31) == 0) sh[threadIdx.x >> 5] = sq;
  __syncthreads();
  float tot = sh[0] + sh[1];
  float norm = sqrtf(fmaxf(tot, 1e-12f));
  float xs = sinhf(norm) * ms / norm;
  long bn = (long)b * N_ + n;
  attnh[bn * DP_ + 1 + h * DH_ + d] = (_Float16)xs;
  if (threadIdx.x == 0) {
    float c = coshf(norm);
    th2[bn * H_ + h] = c * c;
  }
}

// ---------------- combined time component of concatenated heads ----------------
__global__ void k_attime(_Float16* __restrict__ attnh, const float* __restrict__ th2)
{
  int row = blockIdx.x * blockDim.x + threadIdx.x;
  if (row >= MP_) return;
  if (row < BN_) {
    float s = 0.f;
    for (int h = 0; h < H_; h++) s += th2[(long)row * H_ + h];
    float t = sqrtf(fmaxf(s - (float)(H_ - 1), 1e-8f));
    attnh[(long)row * DP_] = (_Float16)t;
    for (int c = D_; c < DP_; c++) attnh[(long)row * DP_ + c] = (_Float16)0.f;
  } else {
    for (int c = 0; c < DP_; c++) attnh[(long)row * DP_ + c] = (_Float16)0.f;
  }
}

// ---------------- residual + lorentz layernorm 2 ----------------
__global__ __launch_bounds__(256)
void k_resln2(const float* __restrict__ of, const float* __restrict__ x,
              const float* __restrict__ gamma, const float* __restrict__ beta,
              float* __restrict__ outs, _Float16* __restrict__ hh)
{
  __shared__ float sh[16];
  int row = blockIdx.x;
  _Float16* hrow = hh + (long)row * DP_;
  if (row >= BN_) { for (int c = threadIdx.x; c < DP_; c += 256) hrow[c] = (_Float16)0.f; return; }
  const float* orow = of + (long)row * DP_ + 1;
  const float* xrow = x + (long)row * D_ + 1;
  float* osr = outs + (long)row * S_;
  float s = 0.f, s2 = 0.f;
  for (int c = threadIdx.x; c < S_; c += 256) { float v = orow[c] + xrow[c]; osr[c] = v; s += v; s2 += v * v; }
  s = blockReduceSum(s, sh);
  s2 = blockReduceSum(s2, sh);
  float mu = s / S_, var = s2 / S_ - mu * mu;
  float inv = rsqrtf(var + 1e-5f);
  float nsq = 0.f;
  for (int c = threadIdx.x; c < S_; c += 256) { float v = (osr[c] - mu) * inv * gamma[c] + beta[c]; nsq += v * v; }
  nsq = blockReduceSum(nsq, sh);
  float t = sqrtf(nsq + 1.f);
  for (int c = threadIdx.x; c < S_; c += 256) { float v = (osr[c] - mu) * inv * gamma[c] + beta[c]; hrow[1 + c] = (_Float16)v; }
  if (threadIdx.x == 0) hrow[0] = (_Float16)t;
  for (int c = S_ + 1 + threadIdx.x; c < DP_; c += 256) hrow[c] = (_Float16)0.f;
}

// ---------------- exact GELU + time component -> f16 FFN activation ----------------
__global__ __launch_bounds__(256)
void k_gelu(const float* __restrict__ g1, _Float16* __restrict__ a2)
{
  __shared__ float sh[16];
  int row = blockIdx.x;
  _Float16* arow = a2 + (long)row * FFP_;
  if (row >= BN_) { for (int c = threadIdx.x; c < FFP_; c += 256) arow[c] = (_Float16)0.f; return; }
  const float* grow = g1 + (long)row * FFP_ + 1;
  float nsq = 0.f;
  for (int c = threadIdx.x; c < MS_; c += 256) {
    float v = grow[c];
    float ge = 0.5f * v * (1.f + erff(v * 0.70710678118f));
    arow[1 + c] = (_Float16)ge;
    nsq += ge * ge;
  }
  nsq = blockReduceSum(nsq, sh);
  if (threadIdx.x == 0) arow[0] = (_Float16)sqrtf(nsq + 1.f);
  for (int c = MS_ + 1 + threadIdx.x; c < FFP_; c += 256) arow[c] = (_Float16)0.f;
}

// ---------------- final residual + add_time -> output ----------------
__global__ __launch_bounds__(256)
void k_final(const float* __restrict__ g2, const float* __restrict__ outs,
             float* __restrict__ out)
{
  __shared__ float sh[16];
  int row = blockIdx.x;
  if (row >= BN_) return;
  const float* grow = g2 + (long)row * DP_ + 1;
  const float* osr = outs + (long)row * S_;
  float* orow = out + (long)row * D_;
  float nsq = 0.f;
  for (int c = threadIdx.x; c < S_; c += 256) { float v = grow[c] + osr[c]; orow[1 + c] = v; nsq += v * v; }
  nsq = blockReduceSum(nsq, sh);
  if (threadIdx.x == 0) orow[0] = sqrtf(nsq + 1.f);
}

// =============================== host ===============================
extern "C" void kernel_launch(void* const* d_in, const int* in_sizes, int n_in,
                              void* d_out, int out_size, void* d_ws, size_t ws_size,
                              hipStream_t stream)
{
  const float* x     = (const float*)d_in[0];
  const float* g1w_  = (const float*)d_in[1];
  const float* b1w_  = (const float*)d_in[2];
  const float* g2w_  = (const float*)d_in[3];
  const float* b2w_  = (const float*)d_in[4];
  const float* Wq    = (const float*)d_in[5];
  const float* Wk    = (const float*)d_in[6];
  const float* Wv    = (const float*)d_in[7];
  const float* Wo    = (const float*)d_in[8];
  const float* W1    = (const float*)d_in[9];
  const float* W2    = (const float*)d_in[10];
  const float* temp  = (const float*)d_in[11];
  const float* araw  = (const float*)d_in[12];
  const float* traw  = (const float*)d_in[13];
  const float* lraw  = (const float*)d_in[14];
  float* out = (float*)d_out;

  // bump allocator over d_ws
  size_t off = 0;
  char* base = (char*)d_ws;
  auto alloc = [&](size_t n) -> void* { off = (off + 255) & ~(size_t)255; void* p = base + off; off += n; return p; };

  _Float16* xnh  = (_Float16*)alloc((size_t)MP_ * DP_ * 2);
  _Float16* Wqh  = (_Float16*)alloc((size_t)DP_ * DP_ * 2);
  _Float16* Wkh  = (_Float16*)alloc((size_t)DP_ * DP_ * 2);
  _Float16* Wvh  = (_Float16*)alloc((size_t)DP_ * DP_ * 2);
  _Float16* Woh  = (_Float16*)alloc((size_t)DP_ * DP_ * 2);
  _Float16* W1h  = (_Float16*)alloc((size_t)FFP_ * DP_ * 2);
  _Float16* W2h  = (_Float16*)alloc((size_t)DP_ * FFP_ * 2);
  float*    qf   = (float*)alloc((size_t)MP_ * DP_ * 4);
  float*    kf   = (float*)alloc((size_t)MP_ * DP_ * 4);
  float*    vf   = (float*)alloc((size_t)MP_ * DP_ * 4);
  _Float16* qh   = (_Float16*)alloc((size_t)BH_ * NPQ_ * DH_ * 2);
  _Float16* kh   = (_Float16*)alloc((size_t)BH_ * NPK_ * DH_ * 2);
  _Float16* vtT  = (_Float16*)alloc((size_t)BH_ * DH_ * NPK_ * 2);
  float*    qt   = (float*)alloc((size_t)BH_ * NPQ_ * 4);
  float*    kt   = (float*)alloc((size_t)BH_ * NPK_ * 4);
  float*    inner= (float*)alloc((size_t)BH_ * NPQ_ * NPK_ * 4);
  _Float16* sch  = (_Float16*)alloc((size_t)BH_ * NPQ_ * NPK_ * 2);
  float*    means= (float*)alloc((size_t)BH_ * NPQ_ * DH_ * 4);
  float*    th2  = (float*)alloc((size_t)BN_ * H_ * 4);
  _Float16* attnh= (_Float16*)alloc((size_t)MP_ * DP_ * 2);
  float*    of   = (float*)alloc((size_t)MP_ * DP_ * 4);
  float*    outs = (float*)alloc((size_t)BN_ * S_ * 4);
  _Float16* hh   = (_Float16*)alloc((size_t)MP_ * DP_ * 2);
  float*    g1b  = (float*)alloc((size_t)MP_ * FFP_ * 4);
  _Float16* a2h  = (_Float16*)alloc((size_t)MP_ * FFP_ * 2);
  float*    g2b  = (float*)alloc((size_t)MP_ * DP_ * 4);
  (void)ws_size; (void)n_in; (void)in_sizes; (void)out_size;

  // 1) pad/convert weights to f16 ([out][in] layout == GEMM "B [n][k]")
  {
    int tot = DP_ * DP_;
    hipLaunchKernelGGL(k_padw, dim3((tot + 255) / 256), dim3(256), 0, stream, Wq, Wqh, D_, D_, DP_, DP_);
    hipLaunchKernelGGL(k_padw, dim3((tot + 255) / 256), dim3(256), 0, stream, Wk, Wkh, D_, D_, DP_, DP_);
    hipLaunchKernelGGL(k_padw, dim3((tot + 255) / 256), dim3(256), 0, stream, Wv, Wvh, D_, D_, DP_, DP_);
    hipLaunchKernelGGL(k_padw, dim3((tot + 255) / 256), dim3(256), 0, stream, Wo, Woh, D_, D_, DP_, DP_);
    int t1 = FFP_ * DP_;
    hipLaunchKernelGGL(k_padw, dim3((t1 + 255) / 256), dim3(256), 0, stream, W1, W1h, M3_, D_, FFP_, DP_);
    int t2 = DP_ * FFP_;
    hipLaunchKernelGGL(k_padw, dim3((t2 + 255) / 256), dim3(256), 0, stream, W2, W2h, D_, M3_, DP_, FFP_);
  }
  // 2) layernorm1 -> xnh (f16 padded)
  hipLaunchKernelGGL(k_ln1, dim3(MP_), dim3(256), 0, stream, x, g1w_, b1w_, xnh);

  // 3) Q/K/V projections via WMMA GEMM
  dim3 gProj(MP_ / 64, DP_ / 32, 1);
  hipLaunchKernelGGL(k_gemm, gProj, dim3(128), 0, stream, xnh, Wqh, qf, DP_, DP_, DP_, DP_, 0L, 0L, 0L);
  hipLaunchKernelGGL(k_gemm, gProj, dim3(128), 0, stream, xnh, Wkh, kf, DP_, DP_, DP_, DP_, 0L, 0L, 0L);
  hipLaunchKernelGGL(k_gemm, gProj, dim3(128), 0, stream, xnh, Wvh, vf, DP_, DP_, DP_, DP_, 0L, 0L, 0L);

  // 4) head split + per-head times + logmap0(v)^T
  hipLaunchKernelGGL(k_heads, dim3((BH_ * NPQ_ + 255) / 256), dim3(256), 0, stream,
                     qf, kf, vf, qh, kh, vtT, qt, kt);

  // 5) inner_s = q_s @ k_s^T per (b,h)
  dim3 gQK(NPQ_ / 64, NPK_ / 32, BH_);
  hipLaunchKernelGGL(k_gemm, gQK, dim3(128), 0, stream, qh, kh, inner, DH_, DH_, NPK_, DH_,
                     (long)NPQ_ * DH_, (long)NPK_ * DH_, (long)NPQ_ * NPK_);

  // 6) hyperbolic logits + softmax
  hipLaunchKernelGGL(k_logits, dim3(NPQ_, BH_), dim3(256), 0, stream,
                     inner, qt, kt, sch, temp, araw, traw, lraw);

  // 7) mean_s = score @ vt (spatial)
  dim3 gAV(NPQ_ / 64, DH_ / 32, BH_);
  hipLaunchKernelGGL(k_gemm, gAV, dim3(128), 0, stream, sch, vtT, means, NPK_, NPK_, DH_, NPK_,
                     (long)NPQ_ * NPK_, (long)DH_ * NPK_, (long)NPQ_ * DH_);

  // 8) expmap0 + concat heads
  hipLaunchKernelGGL(k_expmap, dim3(N_, BH_), dim3(64), 0, stream, means, attnh, th2);
  hipLaunchKernelGGL(k_attime, dim3((MP_ + 255) / 256), dim3(256), 0, stream, attnh, th2);

  // 9) output projection
  hipLaunchKernelGGL(k_gemm, gProj, dim3(128), 0, stream, attnh, Woh, of, DP_, DP_, DP_, DP_, 0L, 0L, 0L);

  // 10) residual + layernorm2
  hipLaunchKernelGGL(k_resln2, dim3(MP_), dim3(256), 0, stream, of, x, g2w_, b2w_, outs, hh);

  // 11) FFN up
  dim3 gUp(MP_ / 64, FFP_ / 32, 1);
  hipLaunchKernelGGL(k_gemm, gUp, dim3(128), 0, stream, hh, W1h, g1b, DP_, DP_, FFP_, DP_, 0L, 0L, 0L);
  hipLaunchKernelGGL(k_gelu, dim3(MP_), dim3(256), 0, stream, g1b, a2h);

  // 12) FFN down
  hipLaunchKernelGGL(k_gemm, gProj, dim3(128), 0, stream, a2h, W2h, g2b, FFP_, FFP_, DP_, FFP_, 0L, 0L, 0L);

  // 13) final residual + add_time
  hipLaunchKernelGGL(k_final, dim3(BN_), dim3(256), 0, stream, g2b, outs, out);
}
